// DistillationLoss_25847113187778
// MI455X (gfx1250) — compile-verified, matrix operand order checked by reading, not run on hardware
//
#include <hip/hip_runtime.h>
#include <hip/hip_bf16.h>
#include <math.h>

#define B_   8
#define AR_  12
#define S_   512
#define NEGV (-1e30f)
#define LOG2E 1.4426950408889634f
#define LN2   0.6931471805599453f

typedef float    v4f  __attribute__((ext_vector_type(4)));
typedef int      v4i  __attribute__((ext_vector_type(4)));
typedef _Float16 v16h __attribute__((ext_vector_type(16)));
typedef float    v8f  __attribute__((ext_vector_type(8)));

// Hardware base-2 transcendentals (v_exp_f32 / v_log_f32 are single TRANS ops)
#if __has_builtin(__builtin_amdgcn_exp2f)
#define EXP2F(x) __builtin_amdgcn_exp2f(x)
#else
#define EXP2F(x) exp2f(x)
#endif
#if __has_builtin(__builtin_amdgcn_logf)
#define LOG2F(x) __builtin_amdgcn_logf(x)
#else
#define LOG2F(x) log2f(x)
#endif

__device__ __forceinline__ float wave_max32(float v) {
#pragma unroll
    for (int off = 16; off >= 1; off >>= 1)
        v = fmaxf(v, __shfl_xor(v, off, 32));
    return v;
}
__device__ __forceinline__ float wave_sum32(float v) {
#pragma unroll
    for (int off = 16; off >= 1; off >>= 1)
        v += __shfl_xor(v, off, 32);
    return v;
}

// One wave32 per row of the (B*AR*S, S) logit matrices.
// Block = 256 threads = 8 waves = 8 consecutive rows.
// Writes one partial sum (sum of elem over the 8 rows) per block.
__global__ __launch_bounds__(256)
void distill_rows_kernel(const float* __restrict__ tlog,
                         const float* __restrict__ slog,
                         const int*   __restrict__ mask,
                         float*       __restrict__ blocksums)
{
    __shared__ float wsum[8];

    const int lane = threadIdx.x & 31;
    const int wave = threadIdx.x >> 5;
    const int row  = blockIdx.x * 8 + wave;
    const int b    = row / (AR_ * S_);
    const int rs   = row & (S_ - 1);

    float elems = 0.0f;
    if (mask[b * S_ + rs] != 0) {           // rowm == 0 -> whole row contributes 0
        const float* trow = tlog + (size_t)row * S_;
        const float* srow = slog + (size_t)row * S_;
        const int*   mrow = mask + b * S_;

        // Load full row into registers (log2 domain): lane covers cols c*128 + lane*4 .. +3
        float tv[16], sv[16];
#pragma unroll
        for (int c = 0; c < 4; ++c) {
            const int col = c * 128 + lane * 4;
            v4f t4 = __builtin_nontemporal_load((const v4f*)(trow + col));
            v4f s4 = __builtin_nontemporal_load((const v4f*)(srow + col));
            v4i m4 = *(const v4i*)(mrow + col);
#pragma unroll
            for (int i = 0; i < 4; ++i) {
                const bool on = (m4[i] != 0);
                tv[c * 4 + i] = (on ? t4[i] : NEGV) * LOG2E;
                sv[c * 4 + i] = (on ? s4[i] : NEGV) * LOG2E;
            }
        }

        // Row max in log2 domain (masked entries are NEGV*LOG2E)
        float tmax = NEGV, smax = NEGV;
#pragma unroll
        for (int k = 0; k < 16; ++k) {
            tmax = fmaxf(tmax, tv[k]);
            smax = fmaxf(smax, sv[k]);
        }
        tmax = wave_max32(tmax);
        smax = wave_max32(smax);

        // Sum of 2^(x - max); masked entries give exp2(-huge) == 0, no branch
        float tsum = 0.0f, ssum = 0.0f;
#pragma unroll
        for (int k = 0; k < 16; ++k) {
            tsum += EXP2F(tv[k] - tmax);
            ssum += EXP2F(sv[k] - smax);
        }
        tsum = wave_sum32(tsum);
        ssum = wave_sum32(ssum);

        const float tls = tmax + LOG2F(tsum);   // log2-sum-exp
        const float sls = smax + LOG2F(ssum);

        // KL sum in log2 domain: 2^a2 * (a2 - c2); natural-log scale applied once per row.
        // Masked cols: a2 == c2 (same constant) -> diff exactly 0, and 2^a2 == 0.
#pragma unroll
        for (int k = 0; k < 16; ++k) {
            const float a2 = tv[k] - tls;       // teacher log2-prob
            const float c2 = sv[k] - sls;       // student log2-prob
            elems += EXP2F(a2) * (a2 - c2);
        }
        elems = wave_sum32(elems) * LN2;        // convert log2 -> ln once per row
    }

    if (lane == 0) wsum[wave] = elems;
    __syncthreads();

    // Reduce the 8 wave sums with one v_wmma_f32_16x16x32_f16.
    // Lane 0's 16 A-slots are all matrix row M=0 (lanes 0-15 hold M=lane),
    // B = all-ones is layout invariant, so D[0][0] = sum of lane0's values.
    // Each f32 partial is split hi/lo into two f16s (error ~2^-22 relative).
    if (wave == 0) {
        v16h a;
#pragma unroll
        for (int i = 0; i < 16; ++i) a[i] = (_Float16)0.0f;
        if (lane == 0) {
#pragma unroll
            for (int i = 0; i < 8; ++i) {
                const float     x  = wsum[i];
                const _Float16  hi = (_Float16)x;
                const _Float16  lo = (_Float16)(x - (float)hi);
                a[2 * i]     = hi;
                a[2 * i + 1] = lo;
            }
        }
        v16h ones;
#pragma unroll
        for (int i = 0; i < 16; ++i) ones[i] = (_Float16)1.0f;
        v8f acc;
#pragma unroll
        for (int i = 0; i < 8; ++i) acc[i] = 0.0f;

        acc = __builtin_amdgcn_wmma_f32_16x16x32_f16(
            /*neg_a=*/false, a, /*neg_b=*/false, ones,
            /*c_mod=*/(short)0, acc, /*reuse_a=*/false, /*reuse_b=*/false);

        if (lane == 0) blocksums[blockIdx.x] = acc[0];  // D[M=0][N=0]
    }
}

// Single-wave deterministic finalize: per-batch sum of 768 block partials,
// per-batch L from the mask, scale by 1/(AR*L^2), mean over batches.
__global__ __launch_bounds__(32)
void distill_finalize_kernel(const float* __restrict__ blocksums,
                             const int*   __restrict__ mask,
                             float*       __restrict__ out)
{
    const int lane = threadIdx.x;  // 32 threads
    float total = 0.0f;
#pragma unroll 1
    for (int b = 0; b < B_; ++b) {
        float p = 0.0f;
        for (int i = lane; i < (AR_ * S_) / 8; i += 32)   // 768 partials per batch
            p += blocksums[b * ((AR_ * S_) / 8) + i];
        p = wave_sum32(p);

        int lc = 0;
        for (int i = lane; i < S_; i += 32)
            lc += mask[b * S_ + i];
#pragma unroll
        for (int off = 16; off >= 1; off >>= 1)
            lc += __shfl_xor(lc, off, 32);

        const float L = (float)lc;
        total += p / ((float)AR_ * L * L);
    }
    if (lane == 0) *out = total / (float)B_;
}

extern "C" void kernel_launch(void* const* d_in, const int* in_sizes, int n_in,
                              void* d_out, int out_size, void* d_ws, size_t ws_size,
                              hipStream_t stream) {
    const float* tlog = (const float*)d_in[0];   // teacher_logits (B,AR,S,S) f32
    const float* slog = (const float*)d_in[1];   // student_logits (B,AR,S,S) f32
    const int*   mask = (const int*)d_in[2];     // attention_mask (B,S) i32
    float*       out  = (float*)d_out;           // scalar f32
    float*       blocksums = (float*)d_ws;       // 6144 floats (24 KB) scratch

    const int nblocks = (B_ * AR_ * S_) / 8;     // 6144 blocks, 8 rows each
    distill_rows_kernel<<<nblocks, 256, 0, stream>>>(tlog, slog, mask, blocksums);
    distill_finalize_kernel<<<1, 32, 0, stream>>>(blocksums, mask, out);
}
